// HieraActionDetector_12902081757296
// MI455X (gfx1250) — compile-verified, matrix operand order checked by reading, not run on hardware
//
#include <hip/hip_runtime.h>
#include <hip/hip_bf16.h>
#include <math.h>

// ---------------- constants from the reference ----------------
#define R_ROIS   1024
#define BATCH    4
#define CH       384
#define NC       21
#define IMG      224
#define POOL     7
#define SR       2
#define DFEAT    (CH * POOL * POOL)   // 18816
#define HID      1024
#define GAMMA_F  2.0f
#define ALPHA_F  0.25f
#define BETA_F   (1.0f / 9.0f)

typedef __attribute__((ext_vector_type(16))) __bf16 v16bf;
typedef __attribute__((ext_vector_type(8)))  __bf16 v8bf;
typedef __attribute__((ext_vector_type(8)))  float  v8f;

static __device__ __forceinline__ unsigned short f2bf(float f) {
  unsigned int u = __float_as_uint(f);
  unsigned int r = (u + 0x7FFFu + ((u >> 16) & 1u)) >> 16;   // RNE
  return (unsigned short)r;
}

// load 16 bf16 for one WMMA operand: two disjoint 16-byte chunks per ISA
// 16-bit A/B layout (lane half hl selects K-base 0/8; chunks at +0 and +16 elems)
static __device__ __forceinline__ v16bf ldv16(const unsigned short* p) {
  v8bf lo = *reinterpret_cast<const v8bf*>(p);
  v8bf hi = *reinterpret_cast<const v8bf*>(p + 16);
  return __builtin_shufflevector(lo, hi, 0,1,2,3,4,5,6,7,8,9,10,11,12,13,14,15);
}

static __device__ __forceinline__ v8f wmma_bf16(v16bf a, v16bf b, v8f c) {
  return __builtin_amdgcn_wmma_f32_16x16x32_bf16(
      false, a, false, b, (short)0, c, false, false);
}

// =====================================================================
// 1) ROIAlign over 4 FPN levels -> bf16 activations [R, 18816]
//    one block per ROI, 256 threads
// =====================================================================
__global__ void roi_align_kernel(const float* __restrict__ p2,
                                 const float* __restrict__ p3,
                                 const float* __restrict__ p4,
                                 const float* __restrict__ p5,
                                 const float* __restrict__ rois,
                                 const int*   __restrict__ batch_idx,
                                 unsigned short* __restrict__ xout) {
  __shared__ int   s_lvl, s_b, s_H;
  __shared__ float s_x1s, s_y1s, s_bw, s_bh, s_Hf;
  __shared__ int   s_x0[POOL * SR], s_x1[POOL * SR], s_y0[POOL * SR], s_y1[POOL * SR];
  __shared__ float s_lx[POOL * SR], s_ly[POOL * SR], s_vx[POOL * SR], s_vy[POOL * SR];

  const int r   = blockIdx.x;
  const int tid = threadIdx.x;

  if (tid == 0) {
    float x1 = rois[r * 4 + 0], y1 = rois[r * 4 + 1];
    float x2 = rois[r * 4 + 2], y2 = rois[r * 4 + 3];
    float area = (x2 - x1) * (y2 - y1);
    float lf = floorf(4.0f + log2f(sqrtf(area) / 224.0f + 1e-6f));
    lf = fminf(fmaxf(lf, 2.0f), 5.0f);
    int lvl = (int)lf - 2;                         // 0..3
    float scale = 0.25f / (float)(1 << lvl);
    int   H = 56 >> lvl;
    float x1s = x1 * scale, y1s = y1 * scale;
    float rw = fmaxf(x2 * scale - x1s, 1.0f);
    float rh = fmaxf(y2 * scale - y1s, 1.0f);
    s_lvl = lvl; s_b = batch_idx[r]; s_H = H; s_Hf = (float)H;
    s_x1s = x1s; s_y1s = y1s;
    s_bw = rw / (float)POOL; s_bh = rh / (float)POOL;
  }
  __syncthreads();

  if (tid < POOL * SR) {
    float k  = ((float)tid + 0.5f) / (float)SR;
    float Hf = s_Hf;
    // x axis
    float xs = s_x1s + s_bw * k;
    s_vx[tid] = (xs >= -1.0f && xs <= Hf) ? 1.0f : 0.0f;
    float x = fminf(fmaxf(xs, 0.0f), Hf - 1.0f);
    float x0f = floorf(x);
    s_lx[tid] = x - x0f;
    int x0 = (int)x0f;
    s_x0[tid] = x0;
    s_x1[tid] = min(x0 + 1, s_H - 1);
    // y axis
    float ys = s_y1s + s_bh * k;
    s_vy[tid] = (ys >= -1.0f && ys <= Hf) ? 1.0f : 0.0f;
    float y = fminf(fmaxf(ys, 0.0f), Hf - 1.0f);
    float y0f = floorf(y);
    s_ly[tid] = y - y0f;
    int y0 = (int)y0f;
    s_y0[tid] = y0;
    s_y1[tid] = min(y0 + 1, s_H - 1);
  }
  __syncthreads();

  const int lvl = s_lvl;
  const int H   = s_H;
  const int b   = s_b;
  const float* feat = (lvl == 0) ? p2 : (lvl == 1) ? p3 : (lvl == 2) ? p4 : p5;

  for (int d = tid; d < DFEAT; d += blockDim.x) {
    int c   = d / (POOL * POOL);
    int rem = d - c * (POOL * POOL);
    int ph  = rem / POOL;
    int pw  = rem - ph * POOL;
    const float* img = feat + ((size_t)(b * CH + c)) * H * H;
    float sum = 0.0f;
    #pragma unroll
    for (int sy = 0; sy < SR; ++sy) {
      int iy = SR * ph + sy;
      float ly = s_ly[iy], hy = 1.0f - ly, vy = s_vy[iy];
      const float* row0 = img + s_y0[iy] * H;
      const float* row1 = img + s_y1[iy] * H;
      #pragma unroll
      for (int sx = 0; sx < SR; ++sx) {
        int ix = SR * pw + sx;
        float lx = s_lx[ix], hx = 1.0f - lx;
        float v = row0[s_x0[ix]] * (hy * hx) + row0[s_x1[ix]] * (hy * lx)
                + row1[s_x0[ix]] * (ly * hx) + row1[s_x1[ix]] * (ly * lx);
        sum += v * (vy * s_vx[ix]);
      }
    }
    xout[(size_t)r * DFEAT + d] = f2bf(sum * 0.25f);
  }
}

// =====================================================================
// 2) Transpose + convert weights: W [K,N] f32 -> WT [N,K] bf16
//    32x32 LDS tiles, block (32,8)
// =====================================================================
__global__ void transpose_bf16_kernel(const float* __restrict__ W,
                                      unsigned short* __restrict__ WT,
                                      int K, int N) {
  __shared__ float tile[32][33];
  int nbase = blockIdx.x * 32;
  int kbase = blockIdx.y * 32;
  for (int j = threadIdx.y; j < 32; j += 8) {
    int k = kbase + j, n = nbase + threadIdx.x;
    if (k < K && n < N) tile[j][threadIdx.x] = W[(size_t)k * N + n];
  }
  __syncthreads();
  for (int j = threadIdx.y; j < 32; j += 8) {
    int n = nbase + j, k = kbase + threadIdx.x;
    if (n < N && k < K) WT[(size_t)n * K + k] = f2bf(tile[threadIdx.x][j]);
  }
}

// =====================================================================
// 3) WMMA GEMM: out = act(A[M,K]bf16 @ BT[N,K]bf16^T + bias)
//    one wave owns a 32M x 64N strip; explicit double-buffered K pipeline:
//    load 6 operands (2A + 4B) for step k0+32, then issue 8 WMMAs for k0.
//    __launch_bounds__(256, 1): allow full VGPR budget (~190 live regs for
//    64 acc + 96 double-buffer operands) so nothing spills to scratch.
//    requires M % 32 == 0, N % 64 == 0 (true here: 1024x1024)
// =====================================================================
__global__ void __launch_bounds__(256, 1)
wmma_gemm_bf16_kernel(const unsigned short* __restrict__ A,
                      const unsigned short* __restrict__ BT,
                      const float* __restrict__ bias,
                      float* __restrict__ outF,
                      unsigned short* __restrict__ outBf,
                      int M, int N, int K, int relu) {
  const int stripsN = N >> 6;                       // strips of 64 cols
  const int totalWaves = (M >> 5) * stripsN;        // strips of 32 rows
  const int wave = blockIdx.x * (blockDim.x >> 5) + (threadIdx.x >> 5);
  if (wave >= totalWaves) return;

  const int tm = wave / stripsN;
  const int tn = wave - tm * stripsN;
  const int lane = threadIdx.x & 31;
  const int lr   = lane & 15;        // row within 16 (A) / col within 16 (B)
  const int hl   = lane >> 4;        // half select
  const int kb   = hl * 8;           // per ISA 16-bit A/B layout

  const unsigned short* pa0 = A  + (size_t)(tm * 32 + lr) * K + kb;
  const unsigned short* pa1 = pa0 + (size_t)16 * K;
  const unsigned short* pb0 = BT + (size_t)(tn * 64 + lr) * K + kb;
  const size_t bs = (size_t)16 * K;

  v8f acc[8] = {};                                  // [mi*4 + nj]

  // prologue: load operands for k0 = 0
  v16bf ca0 = ldv16(pa0);
  v16bf ca1 = ldv16(pa1);
  v16bf cb0 = ldv16(pb0);
  v16bf cb1 = ldv16(pb0 + bs);
  v16bf cb2 = ldv16(pb0 + 2 * bs);
  v16bf cb3 = ldv16(pb0 + 3 * bs);

  // steady state: fetch k0+32 while multiplying k0
  for (int k0 = 0; k0 + 32 < K; k0 += 32) {
    const int kn = k0 + 32;
    __builtin_prefetch(pa0 + kn + 512, 0, 1);
    __builtin_prefetch(pb0 + kn + 512, 0, 1);
    v16bf na0 = ldv16(pa0 + kn);
    v16bf na1 = ldv16(pa1 + kn);
    v16bf nb0 = ldv16(pb0 + kn);
    v16bf nb1 = ldv16(pb0 + bs + kn);
    v16bf nb2 = ldv16(pb0 + 2 * bs + kn);
    v16bf nb3 = ldv16(pb0 + 3 * bs + kn);

    acc[0] = wmma_bf16(ca0, cb0, acc[0]);
    acc[4] = wmma_bf16(ca1, cb0, acc[4]);
    acc[1] = wmma_bf16(ca0, cb1, acc[1]);
    acc[5] = wmma_bf16(ca1, cb1, acc[5]);
    acc[2] = wmma_bf16(ca0, cb2, acc[2]);
    acc[6] = wmma_bf16(ca1, cb2, acc[6]);
    acc[3] = wmma_bf16(ca0, cb3, acc[3]);
    acc[7] = wmma_bf16(ca1, cb3, acc[7]);

    ca0 = na0; ca1 = na1;
    cb0 = nb0; cb1 = nb1; cb2 = nb2; cb3 = nb3;
  }

  // epilogue: final K step
  acc[0] = wmma_bf16(ca0, cb0, acc[0]);
  acc[4] = wmma_bf16(ca1, cb0, acc[4]);
  acc[1] = wmma_bf16(ca0, cb1, acc[1]);
  acc[5] = wmma_bf16(ca1, cb1, acc[5]);
  acc[2] = wmma_bf16(ca0, cb2, acc[2]);
  acc[6] = wmma_bf16(ca1, cb2, acc[6]);
  acc[3] = wmma_bf16(ca0, cb3, acc[3]);
  acc[7] = wmma_bf16(ca1, cb3, acc[7]);

  // C/D layout: lane l -> n = (l&15); vgpr r -> m = r + 8*(l>>4)
  #pragma unroll
  for (int mi = 0; mi < 2; ++mi) {
    #pragma unroll
    for (int j = 0; j < 4; ++j) {
      const int n  = tn * 64 + j * 16 + lr;
      const float bn = bias[n];
      v8f c = acc[mi * 4 + j];
      #pragma unroll
      for (int r = 0; r < 8; ++r) {
        int m = tm * 32 + mi * 16 + r + 8 * hl;
        float v = c[r] + bn;
        if (relu) v = fmaxf(v, 0.0f);
        if (outF)  outF[(size_t)m * N + n]  = v;
        if (outBf) outBf[(size_t)m * N + n] = f2bf(v);
      }
    }
  }
}

// =====================================================================
// 4) cls + bbox heads (tiny): block per row; x row staged into LDS via
//    CDNA5 async global->LDS copy (ASYNCcnt path, §8 async_tensor)
// =====================================================================
__global__ void heads_kernel(const float* __restrict__ X,
                             const float* __restrict__ wc, const float* __restrict__ bc,
                             const float* __restrict__ wb, const float* __restrict__ bb,
                             float* __restrict__ logits, float* __restrict__ breg) {
  __shared__ float xs[HID];
  const int r = blockIdx.x;
  for (int i = threadIdx.x; i < HID; i += blockDim.x) {
    unsigned lofs = (unsigned)(size_t)(&xs[i]);        // LDS byte offset (addr[31:0])
    const float* g = X + (size_t)r * HID + i;
    asm volatile("global_load_async_to_lds_b32 %0, %1, off"
                 :: "v"(lofs), "v"(g) : "memory");
  }
  asm volatile("s_wait_asynccnt 0x0" ::: "memory");
  __syncthreads();
  for (int t = threadIdx.x; t < NC + NC * 4; t += blockDim.x) {
    if (t < NC) {
      float s = bc[t];
      for (int k = 0; k < HID; ++k) s += xs[k] * wc[(size_t)k * NC + t];
      logits[(size_t)r * NC + t] = s;
    } else {
      int j = t - NC;
      float s = bb[j];
      for (int k = 0; k < HID; ++k) s += xs[k] * wb[(size_t)k * (NC * 4) + j];
      breg[(size_t)r * (NC * 4) + j] = s;
    }
  }
}

// =====================================================================
// 5) per-row focal + smooth-L1 terms
// =====================================================================
__global__ void loss_rows_kernel(const float* __restrict__ logits,
                                 const float* __restrict__ breg,
                                 const int*   __restrict__ labels,
                                 const float* __restrict__ regt,
                                 float* __restrict__ terms) {
  int r = blockIdx.x * blockDim.x + threadIdx.x;
  if (r >= R_ROIS) return;
  const float* lg = logits + (size_t)r * NC;
  float mx = lg[0];
  for (int c = 1; c < NC; ++c) mx = fmaxf(mx, lg[c]);
  float se = 0.0f;
  for (int c = 0; c < NC; ++c) se += expf(lg[c] - mx);
  int lab = labels[r];
  float ce = -((lg[lab] - mx) - logf(se));
  float pt = expf(-ce);
  float fw = (1.0f - pt) * (1.0f - pt);            // gamma = 2
  float aw = (lab > 0) ? ALPHA_F : (1.0f - ALPHA_F);
  terms[r * 2 + 0] = fw * aw * ce;

  float box = 0.0f;
  const float* sel = breg + (size_t)r * (NC * 4) + lab * 4;
  #pragma unroll
  for (int i = 0; i < 4; ++i) {
    float d = sel[i] - regt[r * 4 + i];
    float ad = fabsf(d);
    box += (ad < BETA_F) ? (0.5f * d * d / BETA_F) : (ad - 0.5f * BETA_F);
  }
  terms[r * 2 + 1] = (lab > 0) ? box : 0.0f;
}

// =====================================================================
// 6) deterministic single-block reduction -> d_out[2]
// =====================================================================
__global__ void reduce_final_kernel(const float* __restrict__ terms,
                                    float* __restrict__ out) {
  __shared__ float s0[256], s1[256];
  int tid = threadIdx.x;
  float a = 0.0f, b = 0.0f;
  for (int r = tid; r < R_ROIS; r += 256) {   // fixed order per thread
    a += terms[r * 2 + 0];
    b += terms[r * 2 + 1];
  }
  s0[tid] = a; s1[tid] = b;
  __syncthreads();
  for (int off = 128; off > 0; off >>= 1) {
    if (tid < off) { s0[tid] += s0[tid + off]; s1[tid] += s1[tid + off]; }
    __syncthreads();
  }
  if (tid == 0) {
    out[0] = s0[0] / (float)R_ROIS;
    out[1] = s1[0] / (float)R_ROIS;
  }
}

// =====================================================================
// launch
// =====================================================================
extern "C" void kernel_launch(void* const* d_in, const int* in_sizes, int n_in,
                              void* d_out, int out_size, void* d_ws, size_t ws_size,
                              hipStream_t stream) {
  const float* p2   = (const float*)d_in[0];
  const float* p3   = (const float*)d_in[1];
  const float* p4   = (const float*)d_in[2];
  const float* p5   = (const float*)d_in[3];
  const float* rois = (const float*)d_in[4];
  const int*   bidx = (const int*)  d_in[5];
  const int*   labels = (const int*)d_in[6];
  const float* regt = (const float*)d_in[7];
  const float* w6   = (const float*)d_in[8];
  const float* b6   = (const float*)d_in[9];
  const float* w7   = (const float*)d_in[10];
  const float* b7   = (const float*)d_in[11];
  const float* wc   = (const float*)d_in[12];
  const float* bc   = (const float*)d_in[13];
  const float* wb   = (const float*)d_in[14];
  const float* bb   = (const float*)d_in[15];
  float* out = (float*)d_out;

  // workspace carve-out (256B aligned)
  char* ws = (char*)d_ws;
  size_t off = 0;
  auto carve = [&](size_t bytes) -> char* {
    char* p = ws + off;
    off += (bytes + 255) & ~(size_t)255;
    return p;
  };
  unsigned short* x_bf  = (unsigned short*)carve((size_t)R_ROIS * DFEAT * 2); // 38.5 MB
  unsigned short* w6T   = (unsigned short*)carve((size_t)HID * DFEAT * 2);    // 38.5 MB
  unsigned short* h6_bf = (unsigned short*)carve((size_t)R_ROIS * HID * 2);   // 2 MB
  unsigned short* w7T   = (unsigned short*)carve((size_t)HID * HID * 2);      // 2 MB
  float*          h7_f  = (float*)carve((size_t)R_ROIS * HID * 4);            // 4 MB
  float*          logits= (float*)carve((size_t)R_ROIS * NC * 4);
  float*          breg  = (float*)carve((size_t)R_ROIS * NC * 4 * 4);
  float*          terms = (float*)carve((size_t)R_ROIS * 2 * 4);

  // 1) ROIAlign -> bf16 activations
  roi_align_kernel<<<R_ROIS, 256, 0, stream>>>(p2, p3, p4, p5, rois, bidx, x_bf);

  // 2) weight transpose+convert
  {
    dim3 blk(32, 8);
    dim3 g6(HID / 32, DFEAT / 32);   // N=1024, K=18816
    transpose_bf16_kernel<<<g6, blk, 0, stream>>>(w6, w6T, DFEAT, HID);
    dim3 g7(HID / 32, HID / 32);
    transpose_bf16_kernel<<<g7, blk, 0, stream>>>(w7, w7T, HID, HID);
  }

  // 3) fc6: relu(x @ w6 + b6) -> bf16 [1024,1024]
  {
    int waves  = (R_ROIS / 32) * (HID / 64);       // 512 waves
    int blocks = (waves + 7) / 8;                  // 8 waves/block
    wmma_gemm_bf16_kernel<<<blocks, 256, 0, stream>>>(
        x_bf, w6T, b6, nullptr, h6_bf, R_ROIS, HID, DFEAT, 1);
  }

  // 4) fc7: relu(h6 @ w7 + b7) -> f32 [1024,1024]
  {
    int waves  = (R_ROIS / 32) * (HID / 64);
    int blocks = (waves + 7) / 8;
    wmma_gemm_bf16_kernel<<<blocks, 256, 0, stream>>>(
        h6_bf, w7T, b7, h7_f, nullptr, R_ROIS, HID, HID, 1);
  }

  // 5) heads
  heads_kernel<<<R_ROIS, 128, 0, stream>>>(h7_f, wc, bc, wb, bb, logits, breg);

  // 6) losses
  loss_rows_kernel<<<(R_ROIS + 255) / 256, 256, 0, stream>>>(logits, breg, labels, regt, terms);
  reduce_final_kernel<<<1, 256, 0, stream>>>(terms, out);

  (void)in_sizes; (void)n_in; (void)out_size; (void)ws_size;
}
